// Net_39135742001869
// MI455X (gfx1250) — compile-verified, matrix operand order checked by reading, not run on hardware
//
#include <hip/hip_runtime.h>
#include <math.h>

// ---------------- problem constants ----------------
#define Hh    256
#define Bsz   256
#define TENC  80
#define FEAT  4096
#define LCAP  40
#define VOC   6000
#define VPAD  6016   // 94 * 64

typedef __attribute__((ext_vector_type(16))) __bf16 v16bf;
typedef __attribute__((ext_vector_type(8)))  float  v8f;
typedef __attribute__((ext_vector_type(4)))  int    i4v;

// address-space-qualified int4 pointers for the async-to-LDS builtin
typedef __attribute__((address_space(1))) i4v g_i4v;   // global
typedef __attribute__((address_space(3))) i4v l_i4v;   // LDS

// ---------------------------------------------------------------------------
// CDNA5 async global->LDS copy (ASYNCcnt-tracked), with synchronous fallback.
// ---------------------------------------------------------------------------
#if __has_builtin(__builtin_amdgcn_global_load_async_to_lds_b128)
#define HAS_ASYNC_LDS 1
#else
#define HAS_ASYNC_LDS 0
#endif

static __device__ __forceinline__ void async_copy16(void* lds_ptr, const void* gptr) {
#if HAS_ASYNC_LDS
  // integer-detour casts: global generic ptr bits == addrspace(1) bits;
  // low 32 bits of a generic LDS pointer are the LDS byte offset.
  g_i4v* gp = (g_i4v*)(unsigned long long)(__SIZE_TYPE__)gptr;
  l_i4v* lp = (l_i4v*)(unsigned int)(__SIZE_TYPE__)lds_ptr;
  __builtin_amdgcn_global_load_async_to_lds_b128(gp, lp, 0, 0);
#else
  *(uint4*)lds_ptr = *(const uint4*)gptr;
#endif
}

template <int N>
static __device__ __forceinline__ void wait_async() {
#if HAS_ASYNC_LDS
#if __has_builtin(__builtin_amdgcn_s_wait_asynccnt)
  __builtin_amdgcn_s_wait_asynccnt(N);
#else
  asm volatile("s_wait_asynccnt %0" ::"i"(N) : "memory");
#endif
#endif
}

// ---------------------------------------------------------------------------
// WMMA fragment helpers (CDNA5 wave32, v_wmma_f32_16x16x32_bf16)
// A fragment (16x32, row-major source, stride lda):
//   lane L holds row m0+(L&15); K-halves: kb = k0 + (L>>4)*8
//   elements 0..7  -> K = kb..kb+7 ; elements 8..15 -> K = kb+16..kb+23
// B fragment from row-major W[N,K] (column n of B == row n of W): same pattern.
// C/D: VGPR r, lane L -> row m0 + r + 8*(L>>4), col n0 + (L&15)
// ---------------------------------------------------------------------------
static __device__ __forceinline__ v16bf load_frag(const __bf16* __restrict__ P,
                                                  int ld, int row0, int k0, int lane) {
  const __bf16* p = P + (size_t)(row0 + (lane & 15)) * (size_t)ld + (size_t)(k0 + ((lane >> 4) << 3));
  v16bf f;
#pragma unroll
  for (int e = 0; e < 8; ++e) { f[e] = p[e]; f[e + 8] = p[e + 16]; }
  return f;
}

// fragment from LDS tile laid out [rows][32 K] bf16 (row pitch 32 elements)
static __device__ __forceinline__ v16bf load_frag_lds(const __bf16* base, int row0, int lane) {
  const __bf16* p = base + (row0 + (lane & 15)) * 32 + ((lane >> 4) << 3);
  v16bf f;
#pragma unroll
  for (int e = 0; e < 8; ++e) { f[e] = p[e]; f[e + 8] = p[e + 16]; }
  return f;
}

static __device__ __forceinline__ v8f wmma_bf16(v16bf a, v16bf b, v8f c) {
  return __builtin_amdgcn_wmma_f32_16x16x32_bf16(false, a, false, b, (short)0, c, false, false);
}

// ---------------------------------------------------------------------------
// Big hoisted GEMM:  C[M,1024] = A[M,4096] @ W[1024,4096]^T + bias
// Block = 8 waves covering 128 M-rows of one 256-wide N band.
// Wave tile = 16(M) x 256(N): one A fragment feeds 16 WMMAs (4x less A traffic
// than 16x64 tiles).  B tile (256 rows x 32 K = 16 KB) is double-buffered in
// LDS via CDNA5 async-to-LDS loads and shared by all 8 waves.  B fragments are
// software-pipelined (cur/next) so each ds_load has a WMMA of latency cover.
// grid.x = nBlocksM * (1024/256), blockDim = 256
// ---------------------------------------------------------------------------
__global__ void __launch_bounds__(256)
gemm_big(float* __restrict__ C,
         const float* __restrict__ bias,
         const __bf16* __restrict__ A,   // [M,4096]
         const __bf16* __restrict__ W,   // [1024,4096]
         int nBlocksM)
{
  __shared__ __bf16 smem[2][256 * 32];   // 2 x 16 KB

  const int tid  = threadIdx.x;
  const int lane = tid & 31;
  const int wave = tid >> 5;
  const int bm   = blockIdx.x % nBlocksM;
  const int band = blockIdx.x / nBlocksM;
  const int m0   = (bm * 8 + wave) << 4;
  const int n0   = band << 8;             // N band start (256 wide)
  const int rrow = (lane >> 4) << 3;
  const int ncol = lane & 15;

  // stage one 256x32 B tile: thread tid copies the 64 bytes of W row (n0+tid)
  const __bf16* wrow = W + (size_t)(n0 + tid) * FEAT;
  auto stage = [&](int buf, int k) {
    const __bf16* src = wrow + k;
    __bf16* dst = &smem[buf][tid * 32];
    async_copy16(dst,      src);
    async_copy16(dst + 8,  src + 8);
    async_copy16(dst + 16, src + 16);
    async_copy16(dst + 24, src + 24);
  };

  v8f acc[16];
#pragma unroll
  for (int j = 0; j < 16; ++j) {
    const float bv = bias[n0 + (j << 4) + ncol];
#pragma unroll
    for (int r = 0; r < 8; ++r) acc[j][r] = bv;
  }

  stage(0, 0);                                     // prologue: buffer 0
  const __bf16* arow = A + (size_t)(m0 + (lane & 15)) * FEAT + ((lane >> 4) << 3);

  for (int kb = 0; kb < FEAT / 32; ++kb) {
    const int buf = kb & 1;
    if (kb + 1 < FEAT / 32) {
      stage(buf ^ 1, (kb + 1) * 32);               // prefetch next tile
      wait_async<4>();                             // this tile's 4 ops done
    } else {
      wait_async<0>();
    }
    __syncthreads();                               // tile visible to all waves

    // prefetch A one K-step ahead (streams from HBM)
    __builtin_prefetch(arow + (kb + 1) * 32, 0, 0);

    const v16bf afrag = load_frag(A, FEAT, m0, kb * 32, lane);
    const __bf16* bt = &smem[buf][0];

    // software-pipelined B fragments: issue ds_load(j+1) before wmma(j)
    v16bf bcur = load_frag_lds(bt, 0, lane);
#pragma unroll
    for (int j = 0; j < 16; ++j) {
      v16bf bnext = bcur;
      if (j < 15) bnext = load_frag_lds(bt, (j + 1) << 4, lane);
      acc[j] = wmma_bf16(afrag, bcur, acc[j]);
      bcur = bnext;
    }
    __syncthreads();                               // reads done before rewrite
  }

#pragma unroll
  for (int j = 0; j < 16; ++j)
#pragma unroll
    for (int r = 0; r < 8; ++r)
      C[(size_t)(m0 + r + rrow) * 1024 + (size_t)(n0 + (j << 4) + ncol)] = acc[j][r];
}

// ---------------------------------------------------------------------------
// Fused small GEMM for recurrent steps:
//   C[M,N] = init + sum_i A_i[M,K_i] @ W_i[N,K_i]^T
//   init = cinit[row,col] (fp32, stride ldcin) if cinit != nullptr, else bias[col]
// One wave computes a 16(M) x 64(N) tile.  These GEMMs are tiny (K<=768) and
// chained by the recurrence, so launch latency dominates; keep them direct,
// but pipeline the B fragment loads one deep.
// grid: totalWaves = (M/16)*(N/64), blockDim = 256 (8 waves)
// ---------------------------------------------------------------------------
static __device__ __forceinline__ void gemm_pairs(
    v8f (&acc)[4], int m0, int n0, int lane,
    const __bf16* __restrict__ A, int lda,
    const __bf16* __restrict__ W, int ldw, int K)
{
  for (int k = 0; k < K; k += 32) {
    const v16bf a = load_frag(A, lda, m0, k, lane);
    v16bf bcur = load_frag(W, ldw, n0, k, lane);
#pragma unroll
    for (int j = 0; j < 4; ++j) {
      v16bf bnext = bcur;
      if (j < 3) bnext = load_frag(W, ldw, n0 + ((j + 1) << 4), k, lane);
      acc[j] = wmma_bf16(a, bcur, acc[j]);
      bcur = bnext;
    }
  }
}

__global__ void gemm_acc_bf16(
    float* __restrict__ C, int ldc, int nTiles,
    const float* __restrict__ cinit, int ldcin,
    const float* __restrict__ bias,
    const __bf16* __restrict__ A0, int lda0, const __bf16* __restrict__ W0, int ldw0, int K0,
    const __bf16* __restrict__ A1, int lda1, const __bf16* __restrict__ W1, int ldw1, int K1,
    const __bf16* __restrict__ A2, int lda2, const __bf16* __restrict__ W2, int ldw2, int K2)
{
  const int lane  = threadIdx.x & 31;
  const int wid   = (blockIdx.x * blockDim.x + threadIdx.x) >> 5;
  const int mTile = wid / nTiles;
  const int nTile = wid - mTile * nTiles;
  const int m0    = mTile << 4;
  const int n0    = nTile << 6;
  const int rrow  = (lane >> 4) << 3;
  const int ncol  = lane & 15;

  v8f acc[4];
  if (cinit) {
#pragma unroll
    for (int j = 0; j < 4; ++j)
#pragma unroll
      for (int r = 0; r < 8; ++r)
        acc[j][r] = cinit[(size_t)(m0 + r + rrow) * (size_t)ldcin + (size_t)(n0 + (j << 4) + ncol)];
  } else {
#pragma unroll
    for (int j = 0; j < 4; ++j) {
      const float bv = bias[n0 + (j << 4) + ncol];
#pragma unroll
      for (int r = 0; r < 8; ++r) acc[j][r] = bv;
    }
  }

  if (A0) gemm_pairs(acc, m0, n0, lane, A0, lda0, W0, ldw0, K0);
  if (A1) gemm_pairs(acc, m0, n0, lane, A1, lda1, W1, ldw1, K1);
  if (A2) gemm_pairs(acc, m0, n0, lane, A2, lda2, W2, ldw2, K2);

#pragma unroll
  for (int j = 0; j < 4; ++j)
#pragma unroll
    for (int r = 0; r < 8; ++r)
      C[(size_t)(m0 + r + rrow) * (size_t)ldc + (size_t)(n0 + (j << 4) + ncol)] = acc[j][r];
}

// ---------------------------------------------------------------------------
// LSTM elementwise: gates [B,4H] (i,f,g,o) -> c (fp32, in-place), h (bf16)
// ---------------------------------------------------------------------------
static __device__ __forceinline__ float sigf(float x) { return 1.0f / (1.0f + __expf(-x)); }

__global__ void lstm_ew(const float* __restrict__ gates,
                        float* __restrict__ c, __bf16* __restrict__ hbf)
{
  const int idx = blockIdx.x * blockDim.x + threadIdx.x;   // 0..65535
  const int b = idx >> 8, n = idx & 255;
  const float* g = gates + (size_t)b * 1024;
  const float i = sigf(g[n]);
  const float f = sigf(g[256 + n]);
  const float gg = tanhf(g[512 + n]);
  const float o = sigf(g[768 + n]);
  const float cn = f * c[idx] + i * gg;
  c[idx] = cn;
  hbf[idx] = (__bf16)(o * tanhf(cn));
}

// ---------------------------------------------------------------------------
// fp32 -> bf16 conversion
// ---------------------------------------------------------------------------
__global__ void cvt_bf16(const float* __restrict__ s, __bf16* __restrict__ d, long n) {
  const long i = (long)blockIdx.x * blockDim.x + threadIdx.x;
  if (i < n) d[i] = (__bf16)s[i];
}

__global__ void cvt_wout(const float* __restrict__ s, __bf16* __restrict__ d) {
  const int i = blockIdx.x * blockDim.x + threadIdx.x;   // over VPAD*256
  const int r = i >> 8;
  d[i] = (r < VOC) ? (__bf16)s[i] : (__bf16)0.0f;
}

// ---------------------------------------------------------------------------
// prep: zero states/out, fuse biases, pad output bias
// ---------------------------------------------------------------------------
__global__ void prep_kernel(float* c1, float* c2, __bf16* h1bf, __bf16* h2bf,
                            const float* b_ih1, const float* b_hh1, float* b1sum,
                            const float* b_ih2, const float* b_hh2, float* b2sum,
                            const float* b_out, float* boutpad, float* out)
{
  const int idx = blockIdx.x * blockDim.x + threadIdx.x;
  if (idx < Bsz * Hh) {
    c1[idx] = 0.0f; c2[idx] = 0.0f;
    h1bf[idx] = (__bf16)0.0f; h2bf[idx] = (__bf16)0.0f;
  }
  if (idx < 1024) {
    b1sum[idx] = b_ih1[idx] + b_hh1[idx];
    b2sum[idx] = b_ih2[idx] + b_hh2[idx];
  }
  if (idx < VPAD) boutpad[idx] = (idx < VOC) ? b_out[idx] : 0.0f;
  if (idx == 0) out[0] = 0.0f;
}

// ---------------------------------------------------------------------------
// targets = argmax over vocab of caption_one_hot; one block per (b,l)
// ---------------------------------------------------------------------------
__global__ void argmax_kernel(const float* __restrict__ onehot, int* __restrict__ targets) {
  __shared__ float sv[256];
  __shared__ int   si[256];
  const int bl = blockIdx.x, tid = threadIdx.x;
  const float* row = onehot + (size_t)bl * VOC;
  float bv = -1e30f; int bi = 0;
  for (int j = tid; j < VOC; j += 256) { const float v = row[j]; if (v > bv) { bv = v; bi = j; } }
  sv[tid] = bv; si[tid] = bi; __syncthreads();
  for (int s = 128; s > 0; s >>= 1) {
    if (tid < s && sv[tid + s] > sv[tid]) { sv[tid] = sv[tid + s]; si[tid] = si[tid + s]; }
    __syncthreads();
  }
  if (tid == 0) targets[bl] = si[0];
}

// ---------------------------------------------------------------------------
// per-step NLL: one block per batch row; adds mean-over-batch to *out
// ---------------------------------------------------------------------------
__global__ void nll_kernel(const float* __restrict__ logits, const int* __restrict__ targets,
                           int tcol, float* __restrict__ out)
{
  __shared__ float red[256];
  const int b = blockIdx.x, tid = threadIdx.x;
  const float* row = logits + (size_t)b * VPAD;
  float m = -1e30f;
  for (int j = tid; j < VOC; j += 256) m = fmaxf(m, row[j]);
  red[tid] = m; __syncthreads();
  for (int s = 128; s > 0; s >>= 1) { if (tid < s) red[tid] = fmaxf(red[tid], red[tid + s]); __syncthreads(); }
  m = red[0]; __syncthreads();
  float sum = 0.0f;
  for (int j = tid; j < VOC; j += 256) sum += __expf(row[j] - m);
  red[tid] = sum; __syncthreads();
  for (int s = 128; s > 0; s >>= 1) { if (tid < s) red[tid] += red[tid + s]; __syncthreads(); }
  if (tid == 0) {
    const int tgt = targets[b * LCAP + tcol];
    const float nll = -(row[tgt] - m - logf(red[0]));
    atomicAdd(out, nll * (1.0f / (float)Bsz));
  }
}

// ---------------------------------------------------------------------------
// host launcher
// ---------------------------------------------------------------------------
static inline size_t alignup(size_t x) { return (x + 255) & ~(size_t)255; }

extern "C" void kernel_launch(void* const* d_in, const int* in_sizes, int n_in,
                              void* d_out, int out_size, void* d_ws, size_t ws_size,
                              hipStream_t stream)
{
  (void)in_sizes; (void)n_in; (void)out_size; (void)ws_size;
  const float* feat   = (const float*)d_in[0];   // [B,80,4096]
  const float* cap    = (const float*)d_in[1];   // [B,40,256]
  const float* oneh   = (const float*)d_in[2];   // [B,40,6000]
  const float* W_ih1  = (const float*)d_in[3];   // [1024,4096]
  const float* W_hh1  = (const float*)d_in[4];   // [1024,256]
  const float* b_ih1  = (const float*)d_in[5];
  const float* b_hh1  = (const float*)d_in[6];
  const float* W_ih2  = (const float*)d_in[7];   // [1024,512]
  const float* W_hh2  = (const float*)d_in[8];   // [1024,256]
  const float* b_ih2  = (const float*)d_in[9];
  const float* b_hh2  = (const float*)d_in[10];
  const float* W_out  = (const float*)d_in[11];  // [6000,256]
  const float* b_out  = (const float*)d_in[12];
  float* out = (float*)d_out;

  // ---- workspace carve-up ----
  char* base = (char*)d_ws;
  size_t off = 0;
  auto take = [&](size_t bytes) { char* p = base + off; off = alignup(off + bytes); return p; };

  __bf16* featbf  = (__bf16*)take((size_t)Bsz * TENC * FEAT * 2);
  __bf16* Wih1bf  = (__bf16*)take((size_t)1024 * FEAT * 2);
  __bf16* Whh1bf  = (__bf16*)take((size_t)1024 * 256 * 2);
  __bf16* Wih2bf  = (__bf16*)take((size_t)1024 * 512 * 2);
  __bf16* Whh2bf  = (__bf16*)take((size_t)1024 * 256 * 2);
  __bf16* Woutbf  = (__bf16*)take((size_t)VPAD * 256 * 2);
  __bf16* capbf   = (__bf16*)take((size_t)Bsz * LCAP * Hh * 2);
  float*  X1pre   = (float*)take((size_t)Bsz * TENC * 1024 * 4);   // feat@W_ih1^T + b1sum
  float*  b1sum   = (float*)take(1024 * 4);
  float*  b2sum   = (float*)take(1024 * 4);
  float*  boutpad = (float*)take(VPAD * 4);
  __bf16* h1bf    = (__bf16*)take((size_t)Bsz * Hh * 2);
  __bf16* h2bf    = (__bf16*)take((size_t)Bsz * Hh * 2);
  float*  c1      = (float*)take((size_t)Bsz * Hh * 4);
  float*  c2      = (float*)take((size_t)Bsz * Hh * 4);
  float*  gates1  = (float*)take((size_t)Bsz * 1024 * 4);
  float*  gates2  = (float*)take((size_t)Bsz * 1024 * 4);
  float*  logits  = (float*)take((size_t)Bsz * VPAD * 4);
  int*    targets = (int*)take((size_t)Bsz * LCAP * 4);

  // ---- prep + conversions ----
  prep_kernel<<<256, 256, 0, stream>>>(c1, c2, h1bf, h2bf,
                                       b_ih1, b_hh1, b1sum,
                                       b_ih2, b_hh2, b2sum,
                                       b_out, boutpad, out);
  {
    long n = (long)Bsz * TENC * FEAT;
    cvt_bf16<<<(unsigned)((n + 255) / 256), 256, 0, stream>>>(feat, featbf, n);
  }
  cvt_bf16<<<(1024 * FEAT) / 256, 256, 0, stream>>>(W_ih1, Wih1bf, (long)1024 * FEAT);
  cvt_bf16<<<(1024 * 256) / 256, 256, 0, stream>>>(W_hh1, Whh1bf, (long)1024 * 256);
  cvt_bf16<<<(1024 * 512) / 256, 256, 0, stream>>>(W_ih2, Wih2bf, (long)1024 * 512);
  cvt_bf16<<<(1024 * 256) / 256, 256, 0, stream>>>(W_hh2, Whh2bf, (long)1024 * 256);
  cvt_bf16<<<(Bsz * LCAP * Hh) / 256, 256, 0, stream>>>(cap, capbf, (long)Bsz * LCAP * Hh);
  cvt_wout<<<(VPAD * 256) / 256, 256, 0, stream>>>(W_out, Woutbf);
  argmax_kernel<<<Bsz * LCAP, 256, 0, stream>>>(oneh, targets);

  // ---- hoisted big GEMM: X1pre[b,t,:] = feat[b,t,:] @ W_ih1^T + (b_ih1+b_hh1) ----
  // M = 20480 rows (b*80+t), N = 1024, K = 4096.
  // 160 M-blocks (8 waves x 16 rows) x 4 N-bands of 256 -> 640 blocks.
  gemm_big<<<640, 256, 0, stream>>>(X1pre, b1sum, featbf, Wih1bf, 160);

  const int GEMM_BLKS = 32;    // (256/16)*(1024/64)/8 waves per block
  const int LOG_BLKS  = 188;   // (256/16)*(6016/64)/8

  // ---- encoder: 80 steps ----
  for (int t = 0; t < TENC; ++t) {
    // gates1 = X1pre[:,t,:] + h1 @ W_hh1^T
    gemm_acc_bf16<<<GEMM_BLKS, 256, 0, stream>>>(
        gates1, 1024, 16, X1pre + (size_t)t * 1024, TENC * 1024, nullptr,
        h1bf, Hh, Whh1bf, Hh, Hh,
        nullptr, 0, nullptr, 0, 0,
        nullptr, 0, nullptr, 0, 0);
    lstm_ew<<<256, 256, 0, stream>>>(gates1, c1, h1bf);
    // gates2 = b2sum + h1 @ W_ih2[:,256:]^T + h2 @ W_hh2^T   (pad2 half is zero)
    gemm_acc_bf16<<<GEMM_BLKS, 256, 0, stream>>>(
        gates2, 1024, 16, nullptr, 0, b2sum,
        h1bf, Hh, Wih2bf + 256, 512, Hh,
        h2bf, Hh, Whh2bf, Hh, Hh,
        nullptr, 0, nullptr, 0, 0);
    lstm_ew<<<256, 256, 0, stream>>>(gates2, c2, h2bf);
  }

  // ---- decoder: 39 steps ----
  for (int t = 0; t < LCAP - 1; ++t) {
    // lstm1 input is all zeros: gates1 = b1sum + h1 @ W_hh1^T
    gemm_acc_bf16<<<GEMM_BLKS, 256, 0, stream>>>(
        gates1, 1024, 16, nullptr, 0, b1sum,
        h1bf, Hh, Whh1bf, Hh, Hh,
        nullptr, 0, nullptr, 0, 0,
        nullptr, 0, nullptr, 0, 0);
    lstm_ew<<<256, 256, 0, stream>>>(gates1, c1, h1bf);
    // gates2 = b2sum + cap_t @ W_ih2[:,:256]^T + h1 @ W_ih2[:,256:]^T + h2 @ W_hh2^T
    gemm_acc_bf16<<<GEMM_BLKS, 256, 0, stream>>>(
        gates2, 1024, 16, nullptr, 0, b2sum,
        capbf + (size_t)t * Hh, LCAP * Hh, Wih2bf, 512, Hh,
        h1bf, Hh, Wih2bf + 256, 512, Hh,
        h2bf, Hh, Whh2bf, Hh, Hh);
    lstm_ew<<<256, 256, 0, stream>>>(gates2, c2, h2bf);
    // logits = b_out + h2 @ W_out^T  (padded to 6016 cols)
    gemm_acc_bf16<<<LOG_BLKS, 256, 0, stream>>>(
        logits, VPAD, 94, nullptr, 0, boutpad,
        h2bf, Hh, Woutbf, Hh, Hh,
        nullptr, 0, nullptr, 0, 0,
        nullptr, 0, nullptr, 0, 0);
    nll_kernel<<<Bsz, 256, 0, stream>>>(logits, targets, t + 1, out);
  }
}
